// GrokSparseMoeBlock_32985348833414
// MI455X (gfx1250) — compile-verified, matrix operand order checked by reading, not run on hardware
//
#include <hip/hip_runtime.h>
#include <hip/hip_bf16.h>
#include <math.h>

// ---------------- problem constants ----------------
#define Hdim 1024
#define Fdim 4096
#define Edim 8
#define Ttot 4096          // 2 * 2048 tokens
#define MT   32            // tokens per expert-tile (M)
#define FC   256           // F chunk per block (32 per wave)
#define TILES_PER_E (Ttot / MT)   // 128 (worst case: all tokens on one expert)

typedef __attribute__((ext_vector_type(16))) __bf16 v16bf;
typedef __attribute__((ext_vector_type(8)))  __bf16 v8bf;
typedef __attribute__((ext_vector_type(8)))  float  v8f;
typedef __attribute__((ext_vector_type(4)))  unsigned int v4u;
typedef __attribute__((ext_vector_type(8)))  int    v8i;
typedef __attribute__((ext_vector_type(4)))  int    v4i;

// ---------------- workspace layout ----------------
#define WS_IDX_OFF   256
#define WS_WT_OFF    (WS_IDX_OFF + Edim * Ttot * 4)
#define WS_WBF_OFF   ((WS_WT_OFF + Edim * Ttot * 4 + 511) & ~511)          // 262656
#define NW_ELEMS     ((long)Edim * Fdim * Hdim)                            // 33554432 per matrix
#define WS_XBF_OFF   (WS_WBF_OFF + 3L * NW_ELEMS * 2)
#define WS_NEED      (WS_XBF_OFF + (long)Ttot * Hdim * 2)

// D = A(16x32 bf16) * B(32x16 bf16) + C(16x16 f32)
__device__ __forceinline__ v8f wmma_bf16(v16bf a, v16bf b, v8f c) {
    return __builtin_amdgcn_wmma_f32_16x16x32_bf16(
        false, a, false, b, (short)0, c, false, false);
}

// A fragment (16x32 bf16) from LDS, row-major, leading dim `ld`.
// Lane l: row mbase+(l&15); K chunks [koff..koff+7], [koff+16..koff+23], koff=(l<16?0:8).
__device__ __forceinline__ v16bf load_a_lds(const __bf16* base, int mbase, int ld,
                                            int kbase, int lane) {
    const __bf16* p = base + (size_t)(mbase + (lane & 15)) * ld + kbase + ((lane & 16) >> 1);
    v8bf lo = *(const v8bf*)(p);
    v8bf hi = *(const v8bf*)(p + 16);
    v16bf r;
#pragma unroll
    for (int j = 0; j < 8; ++j) { r[j] = lo[j]; r[j + 8] = hi[j]; }
    return r;
}

// B fragment (32x16 bf16) from pre-converted bf16 weights (row-major [outRow][K]).
__device__ __forceinline__ v16bf load_b_frag(const __bf16* __restrict__ W, int nbase,
                                             int ld, int kbase, int lane) {
    const __bf16* p = W + (size_t)(nbase + (lane & 15)) * (size_t)ld + kbase + ((lane & 16) >> 1);
    v8bf lo = *(const v8bf*)(p);
    v8bf hi = *(const v8bf*)(p + 16);
    v16bf r;
#pragma unroll
    for (int j = 0; j < 8; ++j) { r[j] = lo[j]; r[j + 8] = hi[j]; }
    return r;
}

// B fragment built by converting fp32 weights on the fly (fallback path).
__device__ __forceinline__ v16bf load_b_frag(const float* __restrict__ W, int nbase,
                                             int ld, int kbase, int lane) {
    const float* p = W + (size_t)(nbase + (lane & 15)) * (size_t)ld + kbase + ((lane & 16) >> 1);
    float4 a = *(const float4*)(p + 0);
    float4 b = *(const float4*)(p + 4);
    float4 c = *(const float4*)(p + 16);
    float4 d = *(const float4*)(p + 20);
    v16bf r;
    r[0]  = (__bf16)a.x; r[1]  = (__bf16)a.y; r[2]  = (__bf16)a.z; r[3]  = (__bf16)a.w;
    r[4]  = (__bf16)b.x; r[5]  = (__bf16)b.y; r[6]  = (__bf16)b.z; r[7]  = (__bf16)b.w;
    r[8]  = (__bf16)c.x; r[9]  = (__bf16)c.y; r[10] = (__bf16)c.z; r[11] = (__bf16)c.w;
    r[12] = (__bf16)d.x; r[13] = (__bf16)d.y; r[14] = (__bf16)d.z; r[15] = (__bf16)d.w;
    return r;
}

// TDM gather-mode load: 16 rows of `Hdim` elements each, 16-bit row indices.
// D# layout per CDNA5 ISA ch.8 (group0: flags/lds/global+type, group1: dims, groups2/3: indices).
__device__ __forceinline__ void tdm_gather16(unsigned lds_off, const void* gbase,
                                             const int* rows, unsigned data_size_code) {
    unsigned long long ga = (unsigned long long)(uintptr_t)gbase;
    v4u g0;
    g0[0] = 0x80000001u;                              // count=1, gather_mode=1, idx=16-bit
    g0[1] = lds_off;                                  // LDS dest byte address
    g0[2] = (unsigned)(ga & 0xFFFFFFFFu);             // global_addr[31:0]
    g0[3] = (unsigned)((ga >> 32) & 0x01FFFFFFu) | (2u << 30);  // addr[56:32] | type=2
    v8i g1;
    g1[0] = (int)(data_size_code << 16);              // workgroup_mask=0, data_size
    g1[1] = (int)((Hdim & 0xFFFF) << 16);             // tensor_dim0[15:0] @ bits 63:48
    g1[2] = (int)(((Hdim >> 16) & 0xFFFF) | ((Ttot & 0xFFFF) << 16)); // dim0 hi | dim1 lo
    g1[3] = (int)(((Ttot >> 16) & 0xFFFF) | ((Hdim & 0xFFFF) << 16)); // dim1 hi | tile_dim0
    g1[4] = 16;                                       // tile_dim1 = #valid indices
    g1[5] = Hdim;                                     // tensor_dim0_stride low 32
    g1[6] = 0;
    g1[7] = 0;
    v4i g2, g3;
#pragma unroll
    for (int j = 0; j < 4; ++j) {
        g2[j] = (rows[2 * j] & 0xFFFF)     | (rows[2 * j + 1] << 16);
        g3[j] = (rows[8 + 2 * j] & 0xFFFF) | (rows[8 + 2 * j + 1] << 16);
    }
#if __clang_major__ >= 23
    v8i g4 = (v8i)0;
    __builtin_amdgcn_tensor_load_to_lds(g0, g1, g2, g3, g4, 0);
#else
    __builtin_amdgcn_tensor_load_to_lds(g0, g1, g2, g3, 0);
#endif
}

// ---------------- kernel 0: zero counts + output ----------------
__global__ void moe_init_kernel(int* __restrict__ cnt, float* __restrict__ out, int n) {
    int i = blockIdx.x * blockDim.x + threadIdx.x;
    if (i < Edim) cnt[i] = 0;
    if (i < n)    out[i] = 0.0f;
}

// ---------------- kernel 0b: fp32 -> bf16 weight/x pre-conversion ----------------
__global__ __launch_bounds__(256) void moe_convert_kernel(
    const float* __restrict__ wa, const float* __restrict__ wb,
    const float* __restrict__ wc, const float* __restrict__ x,
    __bf16* __restrict__ wbf, __bf16* __restrict__ xbf, long n) {
    long i = ((long)blockIdx.x * blockDim.x + threadIdx.x) * 8;
    long total = 3 * n + (long)Ttot * Hdim;
    if (i >= total) return;
    const float* src;
    __bf16* dst;
    if (i < n)            { src = wa + i;           dst = wbf + i; }
    else if (i < 2 * n)   { src = wb + (i - n);     dst = wbf + i; }
    else if (i < 3 * n)   { src = wc + (i - 2 * n); dst = wbf + i; }
    else                  { src = x + (i - 3 * n);  dst = xbf + (i - 3 * n); }
    float4 v0 = *(const float4*)(src);
    float4 v1 = *(const float4*)(src + 4);
    v8bf o;
    o[0] = (__bf16)v0.x; o[1] = (__bf16)v0.y; o[2] = (__bf16)v0.z; o[3] = (__bf16)v0.w;
    o[4] = (__bf16)v1.x; o[5] = (__bf16)v1.y; o[6] = (__bf16)v1.z; o[7] = (__bf16)v1.w;
    *(v8bf*)dst = o;
}

// ---------------- kernel 1: router (softmax + exact top-2) ----------------
__global__ __launch_bounds__(256) void moe_router_kernel(
    const float* __restrict__ x, const float* __restrict__ gate_w,
    int* __restrict__ cnt, int* __restrict__ idxl, float* __restrict__ wtl) {
    __shared__ float gs[Edim * Hdim];     // 32 KB
    int tid = threadIdx.x;
    for (int i = tid; i < Edim * Hdim; i += blockDim.x) gs[i] = gate_w[i];
    __syncthreads();

    int t = blockIdx.x * blockDim.x + tid;
    if (t >= Ttot) return;
    const float* xp = x + (size_t)t * Hdim;

    float logit[Edim];
#pragma unroll
    for (int e = 0; e < Edim; ++e) logit[e] = 0.0f;
    for (int h = 0; h < Hdim; h += 4) {
        float4 xv = *(const float4*)(xp + h);
#pragma unroll
        for (int e = 0; e < Edim; ++e) {
            logit[e] += xv.x * gs[e * Hdim + h + 0] + xv.y * gs[e * Hdim + h + 1]
                      + xv.z * gs[e * Hdim + h + 2] + xv.w * gs[e * Hdim + h + 3];
        }
    }
    float mx = logit[0];
#pragma unroll
    for (int e = 1; e < Edim; ++e) mx = fmaxf(mx, logit[e]);
    float p[Edim], s = 0.0f;
#pragma unroll
    for (int e = 0; e < Edim; ++e) { p[e] = __expf(logit[e] - mx); s += p[e]; }
    float inv = 1.0f / s;
#pragma unroll
    for (int e = 0; e < Edim; ++e) p[e] *= inv;

    int i1 = 0;
#pragma unroll
    for (int e = 1; e < Edim; ++e) if (p[e] > p[i1]) i1 = e;
    int i2 = (i1 == 0) ? 1 : 0;
#pragma unroll
    for (int e = 0; e < Edim; ++e) if (e != i1 && p[e] > p[i2]) i2 = e;

    int pos1 = atomicAdd(&cnt[i1], 1);
    idxl[i1 * Ttot + pos1] = t;  wtl[i1 * Ttot + pos1] = p[i1];
    int pos2 = atomicAdd(&cnt[i2], 1);
    idxl[i2 * Ttot + pos2] = t;  wtl[i2 * Ttot + pos2] = p[i2];
}

// ---------------- kernel 2: fused sparse SwiGLU expert GEMM ----------------
// grid = E * TILES_PER_E blocks, 256 threads (8 waves).  Wave w owns H-output
// columns [w*128, w*128+128) and F-subchunk [fc + w*32, +32) of each chunk.
// WT = __bf16 : weights/x pre-converted, X tile gathered by TDM (tensor_load_to_lds).
// WT = float  : direct fp32 weights, cooperative gather+convert of X.
template <typename WT>
__global__ __launch_bounds__(256) void moe_expert_kernel(
    const float* __restrict__ x, const __bf16* __restrict__ xb,
    const WT* __restrict__ wlin, const WT* __restrict__ wv, const WT* __restrict__ w1,
    const int* __restrict__ cnt, const int* __restrict__ idxl,
    const float* __restrict__ wtl, float* __restrict__ out) {

    extern __shared__ char smem[];
    __bf16* Xb = (__bf16*)smem;                         // [MT][Hdim]  64 KB
    __bf16* Hs = (__bf16*)(smem + MT * Hdim * 2);       // [MT][FC]    16 KB
    int*   rowTok = (int*)(smem + MT * Hdim * 2 + MT * FC * 2);
    float* rowWt  = (float*)((char*)rowTok + MT * 4);

    const int e    = blockIdx.x >> 7;          // / TILES_PER_E
    const int tile = blockIdx.x & (TILES_PER_E - 1);
    const int nrows = cnt[e] - tile * MT;
    if (nrows <= 0) return;

    const int tid  = threadIdx.x;
    const int wave = tid >> 5;
    const int lane = tid & 31;

    if (tid < MT) {
        int r = tile * MT + tid;
        bool ok = tid < nrows;
        rowTok[tid] = ok ? idxl[e * Ttot + r] : 0;
        rowWt[tid]  = ok ? wtl[e * Ttot + r] : 0.0f;
    }
    __syncthreads();

    if constexpr (sizeof(WT) == 2) {
        // ---- TDM gather: 2 descriptors x 16 rows of bf16 x -> LDS ----
        if (wave == 0) {
            int rows[16];
#pragma unroll
            for (int j = 0; j < 16; ++j) rows[j] = rowTok[j];
            tdm_gather16((unsigned)(uintptr_t)Xb, xb, rows, /*2-byte*/1u);
#pragma unroll
            for (int j = 0; j < 16; ++j) rows[j] = rowTok[16 + j];
            tdm_gather16((unsigned)(uintptr_t)Xb + 16u * Hdim * 2u, xb, rows, 1u);
            __builtin_amdgcn_s_wait_tensorcnt(0);
        }
    } else {
        // ---- fallback: cooperative gather + fp32->bf16 convert ----
        int r  = tid >> 3;
        int c0 = (tid & 7) * 128;
        const float* src = x + (size_t)rowTok[r] * Hdim + c0;
        __bf16* dst = Xb + r * Hdim + c0;
        for (int j = 0; j < 128; j += 4) {
            float4 v = *(const float4*)(src + j);
            dst[j + 0] = (__bf16)v.x; dst[j + 1] = (__bf16)v.y;
            dst[j + 2] = (__bf16)v.z; dst[j + 3] = (__bf16)v.w;
        }
    }
    __syncthreads();

    const WT* Wlin = wlin + (size_t)e * Fdim * Hdim;
    const WT* Wv   = wv   + (size_t)e * Fdim * Hdim;
    const WT* W1   = w1   + (size_t)e * Hdim * Fdim;
    const int ncol0 = wave * 128;              // this wave's H-output base

    v8f acc[2][8];                             // out tile: 32 tokens x 128 cols
#pragma unroll
    for (int mt = 0; mt < 2; ++mt)
#pragma unroll
        for (int nt = 0; nt < 8; ++nt) acc[mt][nt] = (v8f)0.0f;

    for (int fc = 0; fc < Fdim; fc += FC) {
        const int fw = fc + wave * 32;         // this wave's 32 rows of W_lin/W_v

        // ---- GEMM1: g = X*W_lin^T, v = X*W_v^T  (K = Hdim) ----
        v8f g[2][2], vv[2][2];
#pragma unroll
        for (int mt = 0; mt < 2; ++mt)
#pragma unroll
            for (int nt = 0; nt < 2; ++nt) { g[mt][nt] = (v8f)0.0f; vv[mt][nt] = (v8f)0.0f; }

        for (int k = 0; k < Hdim; k += 32) {
            v16bf a0 = load_a_lds(Xb, 0,  Hdim, k, lane);
            v16bf a1 = load_a_lds(Xb, 16, Hdim, k, lane);
#pragma unroll
            for (int nt = 0; nt < 2; ++nt) {
                v16bf bl = load_b_frag(Wlin, fw + nt * 16, Hdim, k, lane);
                g[0][nt]  = wmma_bf16(a0, bl, g[0][nt]);
                g[1][nt]  = wmma_bf16(a1, bl, g[1][nt]);
                v16bf bv = load_b_frag(Wv, fw + nt * 16, Hdim, k, lane);
                vv[0][nt] = wmma_bf16(a0, bv, vv[0][nt]);
                vv[1][nt] = wmma_bf16(a1, bv, vv[1][nt]);
            }
        }

        // ---- SwiGLU: h = silu(g) * v  -> LDS (bf16) ----
        __syncthreads();   // all waves done reading previous Hs chunk
#pragma unroll
        for (int mt = 0; mt < 2; ++mt)
#pragma unroll
            for (int nt = 0; nt < 2; ++nt) {
                v8f gg = g[mt][nt], vvv = vv[mt][nt];
#pragma unroll
                for (int i = 0; i < 8; ++i) {
                    float gvf  = gg[i];
                    float hval = (gvf / (1.0f + __expf(-gvf))) * vvv[i];
                    int m = mt * 16 + i + ((lane & 16) >> 1);     // +8 for lanes 16-31
                    int n = wave * 32 + nt * 16 + (lane & 15);
                    Hs[m * FC + n] = (__bf16)hval;
                }
            }
        __syncthreads();

        __builtin_prefetch(W1 + (size_t)ncol0 * Fdim + fc, 0, 1);

        // ---- GEMM2 rank-update: out += h * W1^T  (K = FC) ----
        for (int k = 0; k < FC; k += 32) {
            v16bf a0 = load_a_lds(Hs, 0,  FC, k, lane);
            v16bf a1 = load_a_lds(Hs, 16, FC, k, lane);
#pragma unroll
            for (int nt = 0; nt < 8; ++nt) {
                v16bf b = load_b_frag(W1, ncol0 + nt * 16, Fdim, fc + k, lane);
                acc[0][nt] = wmma_bf16(a0, b, acc[0][nt]);
                acc[1][nt] = wmma_bf16(a1, b, acc[1][nt]);
            }
        }
    }

    // ---- scale by routing weight, accumulate into d_out ----
#pragma unroll
    for (int mt = 0; mt < 2; ++mt) {
#pragma unroll
        for (int i = 0; i < 8; ++i) {
            int m = mt * 16 + i + ((lane & 16) >> 1);
            int tok = rowTok[m];
            float w = rowWt[m];
            float* op = out + (size_t)tok * Hdim + ncol0 + (lane & 15);
#pragma unroll
            for (int nt = 0; nt < 8; ++nt)
                atomicAdd(op + nt * 16, acc[mt][nt][i] * w);
        }
    }
}

// ---------------- launcher ----------------
extern "C" void kernel_launch(void* const* d_in, const int* in_sizes, int n_in,
                              void* d_out, int out_size, void* d_ws, size_t ws_size,
                              hipStream_t stream) {
    const float* x      = (const float*)d_in[0];   // [2,2048,H]
    const float* gate_w = (const float*)d_in[1];   // [E,H]
    const float* w_lin  = (const float*)d_in[2];   // [E,F,H]
    const float* w_v    = (const float*)d_in[3];   // [E,F,H]
    const float* w_1    = (const float*)d_in[4];   // [E,H,F]
    float* out = (float*)d_out;                    // [2,2048,H]

    char* ws = (char*)d_ws;
    int*   cnt  = (int*)ws;
    int*   idxl = (int*)(ws + WS_IDX_OFF);
    float* wtl  = (float*)(ws + WS_WT_OFF);

    int outN = Ttot * Hdim;
    moe_init_kernel<<<(outN + 255) / 256, 256, 0, stream>>>(cnt, out, outN);
    moe_router_kernel<<<Ttot / 256, 256, 0, stream>>>(x, gate_w, cnt, idxl, wtl);

    size_t smem = (size_t)MT * Hdim * 2 + (size_t)MT * FC * 2 + MT * 4 + MT * 4;

    if (ws_size >= (size_t)WS_NEED) {
        // bf16 fast path: pre-convert weights + x, L2-resident bf16 working set.
        __bf16* wbf = (__bf16*)(ws + WS_WBF_OFF);
        __bf16* xbf = (__bf16*)(ws + WS_XBF_OFF);
        long n = NW_ELEMS;
        long total = 3 * n + (long)Ttot * Hdim;
        int cblocks = (int)((total / 8 + 255) / 256);
        moe_convert_kernel<<<cblocks, 256, 0, stream>>>(w_lin, w_v, w_1, x, wbf, xbf, n);
        moe_expert_kernel<__bf16><<<Edim * TILES_PER_E, 256, smem, stream>>>(
            x, xbf, wbf, wbf + n, wbf + 2 * n, cnt, idxl, wtl, out);
    } else {
        // fallback: stream fp32 weights, convert in-register.
        moe_expert_kernel<float><<<Edim * TILES_PER_E, 256, smem, stream>>>(
            x, (const __bf16*)nullptr, w_lin, w_v, w_1, cnt, idxl, wtl, out);
    }
}
